// PointNetv2Encoder_26877905338408
// MI455X (gfx1250) — matrix operand order repaired and checked by reading, not compiled, not run on hardware
//
#include <hip/hip_runtime.h>
#include <hip/hip_bf16.h>
#include <cstdint>

// ---------------------------------------------------------------------------
// Types for WMMA fragments (gfx1250 wave32, v_wmma_f32_16x16x32_f16)
// ---------------------------------------------------------------------------
typedef __attribute__((ext_vector_type(16))) _Float16 v16h;
typedef __attribute__((ext_vector_type(8)))  _Float16 v8h;
typedef __attribute__((ext_vector_type(8)))  float    v8f;

union Frag16 {
    v16h v;
    v8h  h[2];
};

#define EPS_BN 1e-5f

// ---------------------------------------------------------------------------
// Farthest point sampling: one block (256 threads) per batch element.
// ---------------------------------------------------------------------------
__global__ void fps_kernel(const float* __restrict__ xyz, int N, int npoint,
                           float* __restrict__ nxyz, float* __restrict__ distWS)
{
    const int T = 256;
    __shared__ float sv[8];
    __shared__ int   si[8];
    __shared__ int   sFar;

    int b    = blockIdx.x;
    int tid  = threadIdx.x;
    int lane = tid & 31;
    int wid  = tid >> 5;

    float*       dist = distWS + (size_t)b * N;
    const float* P    = xyz + (size_t)b * N * 3;

    for (int n = tid; n < N; n += T) dist[n] = 1e10f;
    __syncthreads();

    int far = 0;
    for (int j = 0; j < npoint; ++j) {
        float cx = P[(size_t)far * 3 + 0];
        float cy = P[(size_t)far * 3 + 1];
        float cz = P[(size_t)far * 3 + 2];
        if (tid == 0) {
            float* o = nxyz + ((size_t)b * npoint + j) * 3;
            o[0] = cx; o[1] = cy; o[2] = cz;
        }
        float bv = -1.0f;
        int   bi = 0;
        for (int n = tid; n < N; n += T) {
            float dx = P[(size_t)n * 3 + 0] - cx;
            float dy = P[(size_t)n * 3 + 1] - cy;
            float dz = P[(size_t)n * 3 + 2] - cz;
            float d  = dx * dx + dy * dy + dz * dz;
            float dn = fminf(dist[n], d);
            dist[n]  = dn;
            if (dn > bv) { bv = dn; bi = n; }
        }
        for (int off = 16; off > 0; off >>= 1) {
            float ov = __shfl_down(bv, off, 32);
            int   oi = __shfl_down(bi, off, 32);
            if (ov > bv || (ov == bv && oi < bi)) { bv = ov; bi = oi; }
        }
        if (lane == 0) { sv[wid] = bv; si[wid] = bi; }
        __syncthreads();
        if (tid == 0) {
            float mbv = sv[0]; int mbi = si[0];
            for (int t = 1; t < 8; ++t)
                if (sv[t] > mbv || (sv[t] == mbv && si[t] < mbi)) { mbv = sv[t]; mbi = si[t]; }
            sFar = mbi;
        }
        __syncthreads();
        far = sFar;
    }
}

// ---------------------------------------------------------------------------
// Ball query, both radii in one pass. Source points staged via LDS tiles.
// ---------------------------------------------------------------------------
__global__ void ballquery_kernel(const float* __restrict__ xyz,
                                 const float* __restrict__ nxyz,
                                 int N, int S,
                                 float r0sq, int K0, int* __restrict__ g0,
                                 float r1sq, int K1, int* __restrict__ g1)
{
    __shared__ float tx[256], ty[256], tz[256];
    int tid = threadIdx.x;
    int s   = blockIdx.x * 256 + tid;
    int b   = blockIdx.y;

    const float* q = nxyz + ((size_t)b * S + s) * 3;
    float qx = q[0], qy = q[1], qz = q[2];

    int  c0 = 0, c1 = 0;
    int* o0 = g0 + ((size_t)b * S + s) * K0;
    int* o1 = g1 + ((size_t)b * S + s) * K1;

    for (int n0 = 0; n0 < N; n0 += 256) {
        __syncthreads();
        const float* p = xyz + ((size_t)b * N + n0 + tid) * 3;
        tx[tid] = p[0]; ty[tid] = p[1]; tz[tid] = p[2];
        __syncthreads();
        for (int i = 0; i < 256; ++i) {
            float dx = tx[i] - qx, dy = ty[i] - qy, dz = tz[i] - qz;
            float d  = dx * dx + dy * dy + dz * dz;
            if (d <= r1sq) {
                if (c1 < K1) o1[c1++] = n0 + i;
                if (d <= r0sq && c0 < K0) o0[c0++] = n0 + i;
            }
        }
    }
    int f0 = c0 ? o0[0] : 0;
    for (int i = c0; i < K0; ++i) o0[i] = f0;
    int f1 = c1 ? o1[0] : 0;
    for (int i = c1; i < K1; ++i) o1[i] = f1;
}

// ---------------------------------------------------------------------------
// Grouping kernels: build f16 activation rows (zero-padded to K-step stride).
// ---------------------------------------------------------------------------
__global__ void sa1_group_kernel(const float* __restrict__ xyz, int N,
                                 const float* __restrict__ nxyz, int S,
                                 const int* __restrict__ gidx, int K,
                                 _Float16* __restrict__ Xo, int total)
{
    int r = blockIdx.x * blockDim.x + threadIdx.x;
    if (r >= total) return;
    int s = (r / K) % S;
    int b = r / (K * S);
    int g = gidx[r];
    const float* p = xyz  + ((size_t)b * N + g) * 3;
    const float* q = nxyz + ((size_t)b * S + s) * 3;
    _Float16* o = Xo + (size_t)r * 32;
    o[0] = (_Float16)p[0];
    o[1] = (_Float16)p[1];
    o[2] = (_Float16)p[2];
    o[3] = (_Float16)(p[0] - q[0]);
    o[4] = (_Float16)(p[1] - q[1]);
    o[5] = (_Float16)(p[2] - q[2]);
#pragma unroll
    for (int c = 6; c < 32; ++c) o[c] = (_Float16)0.0f;
}

__global__ void sa2_group_kernel(const float* __restrict__ l1xyz,
                                 const _Float16* __restrict__ l1ph, // [B*1024, 96]
                                 const float* __restrict__ l2xyz, int S,
                                 const int* __restrict__ gidx, int K,
                                 _Float16* __restrict__ Xo, int total)
{
    int r = blockIdx.x * blockDim.x + threadIdx.x;
    if (r >= total) return;
    int s = (r / K) % S;
    int b = r / (K * S);
    int g = gidx[r];
    const _Float16* f = l1ph + ((size_t)b * 1024 + g) * 96;
    _Float16* o = Xo + (size_t)r * 128;
#pragma unroll 4
    for (int c = 0; c < 96; ++c) o[c] = f[c];
    const float* p = l1xyz + ((size_t)b * 1024 + g) * 3;
    const float* q = l2xyz + ((size_t)b * S + s) * 3;
    o[96] = (_Float16)(p[0] - q[0]);
    o[97] = (_Float16)(p[1] - q[1]);
    o[98] = (_Float16)(p[2] - q[2]);
#pragma unroll
    for (int c = 99; c < 128; ++c) o[c] = (_Float16)0.0f;
}

// ---------------------------------------------------------------------------
// Weight pre-pack: W [O, C] f32 -> Wh [Opad, Cpad] f16, zero-padded.
// Removes all guards/converts from the GEMM hot loop.
// ---------------------------------------------------------------------------
__global__ void pack_weights_kernel(const float* __restrict__ W, int C, int O,
                                    _Float16* __restrict__ Wh, int Cpad, int total)
{
    int i = blockIdx.x * blockDim.x + threadIdx.x;
    if (i >= total) return;
    int n = i / Cpad;
    int c = i % Cpad;
    Wh[i] = (_Float16)((n < O && c < C) ? W[(size_t)n * C + c] : 0.0f);
}

// Fold bias + BN into per-output scale/shift; pad outputs get (0,0) so the
// f16 store path needs no column guard (pad cols become exactly 0).
__global__ void prep_epi_kernel(const float* __restrict__ b,
                                const float* __restrict__ gamma,
                                const float* __restrict__ beta,
                                const float* __restrict__ mean,
                                const float* __restrict__ var,
                                int O, int Opad,
                                float* __restrict__ sc, float* __restrict__ sh)
{
    int o = blockIdx.x * blockDim.x + threadIdx.x;
    if (o >= Opad) return;
    if (o >= O) { sc[o] = 0.0f; sh[o] = 0.0f; return; }
    if (gamma != nullptr) {
        float s = gamma[o] * rsqrtf(var[o] + EPS_BN);
        sc[o] = s;
        sh[o] = (b[o] - mean[o]) * s + beta[o];
    } else {
        sc[o] = 1.0f;
        sh[o] = b[o];
    }
}

// ---------------------------------------------------------------------------
// WMMA GEMM: Y[R, ldo] = relu?(scale ⊙ (X[R, lda] * Wh^T) + shift).
// One wave owns a 16-row M-tile and all TN = ldo/16 N-tiles (A reuse).
// A frag (f16 16x32): lanes 0-15 row M=l hold K=k0+0..15;
//                     lanes 16-31 hold K=k0+16..31.
// B frag (f16 32x16): lanes 0-15 col N=l hold K=k0+0..15 (contiguous 32B);
//                     lanes 16-31 hold K=k0+16..31.
// ---------------------------------------------------------------------------
template <int TN>
__global__ void gemm_kernel(const _Float16* __restrict__ X, int lda,
                            const _Float16* __restrict__ Wh,
                            const float* __restrict__ sc,
                            const float* __restrict__ sh,
                            int O, int doRelu,
                            _Float16* __restrict__ Yh,
                            float* __restrict__ Yf, int ldo)
{
    int wave = (blockIdx.x * blockDim.x + threadIdx.x) >> 5; // m-tile index
    int lane = threadIdx.x & 31;
    int half = lane >> 4;
    int l    = lane & 15;

    size_t m = (size_t)wave * 16 + l; // A row for this lane

    v8f acc[TN];
#pragma unroll
    for (int t = 0; t < TN; ++t) acc[t] = {};

    for (int k0 = 0; k0 < lda; k0 += 32) {
        Frag16 a;
        const _Float16* xr = X + m * lda + k0;
        a.h[0] = *(const v8h*)(xr + half * 16);
        a.h[1] = *(const v8h*)(xr + half * 16 + 8);
        if (k0 + 32 < lda)
            __builtin_prefetch(xr + 32, 0, 1); // global_prefetch_b8

#pragma unroll
        for (int t = 0; t < TN; ++t) {
            Frag16 bf;
            bf.v = *(const v16h*)(Wh + (size_t)(t * 16 + l) * lda + k0 + half * 16);
            acc[t] = __builtin_amdgcn_wmma_f32_16x16x32_f16(
                false, a.v, false, bf.v, (short)0, acc[t], false, false);
        }
    }

#pragma unroll
    for (int t = 0; t < TN; ++t) {
        int   o = t * 16 + l;
        float s = sc[o];
        float h = sh[o];
        size_t rowBase = (size_t)wave * 16 + half * 8;
#pragma unroll
        for (int i = 0; i < 8; ++i) {
            float v = acc[t][i] * s + h;
            if (doRelu) v = fmaxf(v, 0.0f);
            size_t row = rowBase + i;
            if (Yf != nullptr) {
                if (o < O) Yf[row * ldo + o] = v;
            } else {
                Yh[row * ldo + o] = (_Float16)v; // pad cols: s=h=0 -> 0
            }
        }
    }
}

// ---------------------------------------------------------------------------
// Max-pool over K neighbors: f16 [SB, K, ld] -> f32 (d_out slice) + f16 copy.
// ---------------------------------------------------------------------------
__global__ void maxpool_kernel(const _Float16* __restrict__ Y, int ld, int K, int C,
                               float* __restrict__ outF, int ldo, int colOff,
                               _Float16* __restrict__ outH, int ldh, int total)
{
    int i = blockIdx.x * blockDim.x + threadIdx.x;
    if (i >= total) return;
    int sb = i / C;
    int c  = i % C;
    const _Float16* p = Y + (size_t)sb * K * ld + c;
    float m = -3.4e38f;
    for (int k = 0; k < K; ++k) m = fmaxf(m, (float)p[(size_t)k * ld]);
    outF[(size_t)sb * ldo + colOff + c] = m;
    outH[(size_t)sb * ldh + colOff + c] = (_Float16)m;
}

// feature[:, :, 125:128] = l2_xyz
__global__ void feat_tail_kernel(const float* __restrict__ l2xyz,
                                 float* __restrict__ feat, int total)
{
    int i = blockIdx.x * blockDim.x + threadIdx.x;
    if (i >= total) return;
    int s = i / 3, c = i % 3;
    feat[(size_t)s * 128 + 125 + c] = l2xyz[i];
}

// ---------------------------------------------------------------------------
// Host side
// ---------------------------------------------------------------------------
struct LayerP { const float *w, *b, *gamma, *beta, *mean, *var; };

static inline int cdiv(int a, int b) { return (a + b - 1) / b; }

extern "C" void kernel_launch(void* const* d_in, const int* in_sizes, int n_in,
                              void* d_out, int out_size, void* d_ws, size_t ws_size,
                              hipStream_t stream)
{
    (void)n_in; (void)out_size; (void)ws_size;
    const int B = 4, N0 = 16384, S1 = 1024, S2 = 256;

    auto F = [&](int i) { return (const float*)d_in[i]; };

    // -------- input flatten-order detection (jax-sorted vs insertion) --------
    const float* xyz;
    const float *linW, *linB;
    LayerP L[12]; // 0-2: sa1.b0, 3-5: sa1.b1, 6-8: sa2.b0, 9-11: sa2.b1
    if (in_sizes[0] == B * N0 * 3) {
        xyz = F(0);
        int idx = 1;
        for (int t = 0; t < 12; ++t) {
            L[t].w = F(idx); L[t].b = F(idx + 1); L[t].gamma = F(idx + 2);
            L[t].beta = F(idx + 3); L[t].mean = F(idx + 4); L[t].var = F(idx + 5);
            idx += 6;
        }
        linW = F(73); linB = F(74);
    } else {
        linB = F(0); linW = F(1);
        int idx = 2;
        for (int t = 0; t < 12; ++t) {
            L[t].b = F(idx); L[t].beta = F(idx + 1); L[t].gamma = F(idx + 2);
            L[t].mean = F(idx + 3); L[t].var = F(idx + 4); L[t].w = F(idx + 5);
            idx += 6;
        }
        xyz = F(74);
    }

    // -------- output layout (floats) --------
    float* out = (float*)d_out;
    const size_t OFF_FEAT = 0;                   // (4,256,128)
    const size_t OFF_L0   = OFF_FEAT + 131072;   // (4,16384,3)
    const size_t OFF_L1X  = OFF_L0   + 196608;   // (4,1024,3)
    const size_t OFF_L2X  = OFF_L1X  + 12288;    // (4,256,3)
    const size_t OFF_L1P  = OFF_L2X  + 3072;     // (4,1024,96)
    const size_t OFF_L2P  = OFF_L1P  + 393216;   // (4,256,256)
    float* l1xyz = out + OFF_L1X;
    float* l2xyz = out + OFF_L2X;
    float* l1pts = out + OFF_L1P;
    float* l2pts = out + OFF_L2P;

    // -------- workspace carve-up --------
    char*  wsb = (char*)d_ws;
    size_t cur = 0;
    auto alloc = [&](size_t bytes) -> void* {
        void* p = wsb + cur;
        cur = (cur + bytes + 255) & ~(size_t)255;
        return p;
    };
    float*    distWS = (float*)alloc((size_t)B * N0 * 4);
    int*      g1a    = (int*)alloc((size_t)B * S1 * 16 * 4);
    int*      g1b    = (int*)alloc((size_t)B * S1 * 32 * 4);
    int*      g2a    = (int*)alloc((size_t)B * S2 * 16 * 4);
    int*      g2b    = (int*)alloc((size_t)B * S2 * 32 * 4);
    _Float16* l1ph   = (_Float16*)alloc((size_t)B * S1 * 96 * 2);
    _Float16* l2ph   = (_Float16*)alloc((size_t)B * S2 * 256 * 2);
    _Float16* bufA   = (_Float16*)alloc((size_t)4 * 1024 * 1024 * 2); // 8 MB
    _Float16* bufB   = (_Float16*)alloc((size_t)8 * 1024 * 1024 * 2); // 16 MB

    // -------- GEMM configs (order of use) --------
    // idx:      0   1   2    3   4   5    6   7   8     9  10  11   12(linear)
    const int cfgC[13]   = {  6, 16, 16,   6, 32, 32,  99, 64, 64,  99, 64, 96, 256 };
    const int cfgO[13]   = { 16, 16, 32,  32, 32, 64,  64, 64,128,  64, 96,128, 125 };
    const int cfgLda[13] = { 32, 32, 32,  32, 32, 32, 128, 64, 64, 128, 64, 96, 256 };
    const int cfgLdo[13] = { 32, 32, 32,  32, 32, 64,  64, 64,128,  64, 96,128, 128 };

    LayerP lin = { linW, linB, nullptr, nullptr, nullptr, nullptr };
    _Float16* Wp[13];
    float*    Sc[13];
    float*    Sh[13];
    for (int i = 0; i < 13; ++i) {
        const LayerP& P = (i < 12) ? L[i] : lin;
        int Cpad = cfgLda[i], Opad = cfgLdo[i];
        int wtot = Opad * Cpad;
        Wp[i] = (_Float16*)alloc((size_t)wtot * 2);
        Sc[i] = (float*)alloc((size_t)Opad * 4);
        Sh[i] = (float*)alloc((size_t)Opad * 4);
        pack_weights_kernel<<<cdiv(wtot, 256), 256, 0, stream>>>(
            P.w, cfgC[i], cfgO[i], Wp[i], Cpad, wtot);
        prep_epi_kernel<<<1, Opad, 0, stream>>>(
            P.b, P.gamma, P.beta, P.mean, P.var, cfgO[i], Opad, Sc[i], Sh[i]);
    }

    auto gemm = [&](const _Float16* X, int gi, _Float16* Yh, float* Yf, int R, int relu) {
        int lda = cfgLda[gi], ldo = cfgLdo[gi];
        int tilesM = R / 16;
        dim3 g(tilesM / 8), blk(256);
        switch (ldo / 16) {
        case 2: gemm_kernel<2><<<g, blk, 0, stream>>>(X, lda, Wp[gi], Sc[gi], Sh[gi], cfgO[gi], relu, Yh, Yf, ldo); break;
        case 4: gemm_kernel<4><<<g, blk, 0, stream>>>(X, lda, Wp[gi], Sc[gi], Sh[gi], cfgO[gi], relu, Yh, Yf, ldo); break;
        case 6: gemm_kernel<6><<<g, blk, 0, stream>>>(X, lda, Wp[gi], Sc[gi], Sh[gi], cfgO[gi], relu, Yh, Yf, ldo); break;
        default: gemm_kernel<8><<<g, blk, 0, stream>>>(X, lda, Wp[gi], Sc[gi], Sh[gi], cfgO[gi], relu, Yh, Yf, ldo); break;
        }
    };

    // ===================== pipeline =====================
    hipMemcpyAsync(out + OFF_L0, xyz, (size_t)B * N0 * 3 * 4,
                   hipMemcpyDeviceToDevice, stream);

    // ---- SA1 ----
    fps_kernel<<<dim3(B), dim3(256), 0, stream>>>(xyz, N0, S1, l1xyz, distWS);
    ballquery_kernel<<<dim3(S1 / 256, B), dim3(256), 0, stream>>>(
        xyz, l1xyz, N0, S1, 0.05f * 0.05f, 16, g1a, 0.1f * 0.1f, 32, g1b);

    {   // branch 0: cin 6 -> [16,16,32], K=16
        int R = B * S1 * 16;
        sa1_group_kernel<<<cdiv(R, 256), 256, 0, stream>>>(xyz, N0, l1xyz, S1, g1a, 16, bufA, R);
        gemm(bufA, 0, bufB, nullptr, R, 1);
        gemm(bufB, 1, bufA, nullptr, R, 1);
        gemm(bufA, 2, bufB, nullptr, R, 1);
        int total = B * S1 * 32;
        maxpool_kernel<<<cdiv(total, 256), 256, 0, stream>>>(bufB, 32, 16, 32,
                                                             l1pts, 96, 0, l1ph, 96, total);
    }
    {   // branch 1: cin 6 -> [32,32,64], K=32
        int R = B * S1 * 32;
        sa1_group_kernel<<<cdiv(R, 256), 256, 0, stream>>>(xyz, N0, l1xyz, S1, g1b, 32, bufA, R);
        gemm(bufA, 3, bufB, nullptr, R, 1);
        gemm(bufB, 4, bufA, nullptr, R, 1);
        gemm(bufA, 5, bufB, nullptr, R, 1);
        int total = B * S1 * 64;
        maxpool_kernel<<<cdiv(total, 256), 256, 0, stream>>>(bufB, 64, 32, 64,
                                                             l1pts, 96, 32, l1ph, 96, total);
    }

    // ---- SA2 ----
    fps_kernel<<<dim3(B), dim3(256), 0, stream>>>(l1xyz, S1, S2, l2xyz, distWS);
    ballquery_kernel<<<dim3(S2 / 256, B), dim3(256), 0, stream>>>(
        l1xyz, l2xyz, S1, S2, 0.1f * 0.1f, 16, g2a, 0.2f * 0.2f, 32, g2b);

    {   // branch 0: cin 99 -> [64,64,128], K=16
        int R = B * S2 * 16;
        sa2_group_kernel<<<cdiv(R, 256), 256, 0, stream>>>(l1xyz, l1ph, l2xyz, S2, g2a, 16, bufA, R);
        gemm(bufA, 6, bufB, nullptr, R, 1);
        gemm(bufB, 7, bufA, nullptr, R, 1);
        gemm(bufA, 8, bufB, nullptr, R, 1);
        int total = B * S2 * 128;
        maxpool_kernel<<<cdiv(total, 256), 256, 0, stream>>>(bufB, 128, 16, 128,
                                                             l2pts, 256, 0, l2ph, 256, total);
    }
    {   // branch 1: cin 99 -> [64,96,128], K=32
        int R = B * S2 * 32;
        sa2_group_kernel<<<cdiv(R, 256), 256, 0, stream>>>(l1xyz, l1ph, l2xyz, S2, g2b, 32, bufA, R);
        gemm(bufA,  9, bufB, nullptr, R, 1);
        gemm(bufB, 10, bufA, nullptr, R, 1);
        gemm(bufA, 11, bufB, nullptr, R, 1);
        int total = B * S2 * 128;
        maxpool_kernel<<<cdiv(total, 256), 256, 0, stream>>>(bufB, 128, 32, 128,
                                                             l2pts, 256, 128, l2ph, 256, total);
    }

    // ---- final linear (256 -> 125) into feature cols [0,125), f32 out ----
    gemm(l2ph, 12, nullptr, out + OFF_FEAT, B * S2, 0);

    // feature cols [125,128) = l2_xyz
    feat_tail_kernel<<<cdiv(B * S2 * 3, 256), 256, 0, stream>>>(l2xyz, out + OFF_FEAT, B * S2 * 3);
}
